// SelfAttention_83863531422291
// MI455X (gfx1250) — compile-verified, matrix operand order checked by reading, not run on hardware
//
#include <hip/hip_runtime.h>
#include <hip/hip_bf16.h>
#include <stdint.h>

// ---------------------------------------------------------------------------
// Self-attention for B=2, S=2048, E=1024, H=16, D=64 on gfx1250 (MI455X).
// bf16 WMMA (v_wmma_f32_16x16x32_bf16) for all four matmuls, flash-attention
// streaming softmax, f32 accumulation, and Tensor-Data-Mover (TDM) async
// staging of K/V and Wo tiles into LDS (double buffered, s_wait_tensorcnt).
// ---------------------------------------------------------------------------

typedef __bf16 bf16;
typedef uint32_t u32;
typedef __attribute__((ext_vector_type(16))) __bf16 v16bf;
typedef __attribute__((ext_vector_type(8)))  float  v8f;
typedef __attribute__((ext_vector_type(4)))  u32    v4u;
typedef __attribute__((ext_vector_type(8)))  int    v8i;
typedef __attribute__((ext_vector_type(4)))  int    v4i;

#define B_ 2
#define S_ 2048
#define E_ 1024
#define H_ 16
#define D_ 64

#define KT 32      // keys per flash iteration (one WMMA K-step for P@V)
#define WAVES 8    // waves per block

#if __has_builtin(__builtin_amdgcn_tensor_load_to_lds) && \
    __has_builtin(__builtin_amdgcn_s_wait_tensorcnt)
#define USE_TDM 1
#else
#define USE_TDM 0
#endif

__device__ __forceinline__ int lane_id() { return (int)(threadIdx.x & 31); }

// k-index of element e (0..15) of a 16-bit A/B operand held by `lane`
// (CDNA5 ISA 7.12.2: 16-bit A-matrix 16x32 layout).
__device__ __forceinline__ int amap(int lane, int e) {
  return ((e >> 3) << 4) + ((lane >> 4) << 3) + (e & 7);
}

__device__ __forceinline__ v16bf load_a_f32(const float* __restrict__ src, int ld) {
  int l = lane_id(); int m = l & 15;
  v16bf a;
#pragma unroll
  for (int e = 0; e < 16; ++e) a[e] = (__bf16)src[m * ld + amap(l, e)];
  return a;
}

__device__ __forceinline__ v16bf load_a_bf16(const bf16* __restrict__ src, int ld) {
  int l = lane_id(); int m = l & 15;
  v16bf a;
#pragma unroll
  for (int e = 0; e < 16; ++e) a[e] = src[m * ld + amap(l, e)];
  return a;
}

// B tile 32x16 where B[k][n] = W[n][k]  (W row-major f32, leading dim ld).
__device__ __forceinline__ v16bf load_bt_f32(const float* __restrict__ W, int ld) {
  int l = lane_id(); int n = l & 15;
  v16bf b;
#pragma unroll
  for (int e = 0; e < 16; ++e) b[e] = (__bf16)W[n * ld + amap(l, e)];
  return b;
}

// B tile 32x16 where B[k][n] = W[n][k]  (W row-major bf16, leading dim ld).
__device__ __forceinline__ v16bf load_bt_bf16(const bf16* __restrict__ W, int ld) {
  int l = lane_id(); int n = l & 15;
  v16bf b;
#pragma unroll
  for (int e = 0; e < 16; ++e) b[e] = W[n * ld + amap(l, e)];
  return b;
}

// B tile 32x16 from row-major M[k][n] (bf16, leading dim ld).
__device__ __forceinline__ v16bf load_b_bf16(const bf16* __restrict__ M, int ld) {
  int l = lane_id(); int n = l & 15;
  v16bf b;
#pragma unroll
  for (int e = 0; e < 16; ++e) b[e] = M[amap(l, e) * ld + n];
  return b;
}

#if USE_TDM
// ---- Tensor DMA descriptor packing (CDNA5 ISA ch.8, D# groups 0/1) --------
// Group 0: [1:0]=count(1), [63:32]=lds_addr, [120:64]=global_addr, [127:126]=2.
__device__ __forceinline__ v4u tdm_g0(u32 lds_addr, const void* gaddr) {
  uint64_t ga = (uint64_t)(uintptr_t)gaddr;
  v4u g;
  g[0] = 1u;
  g[1] = lds_addr;
  g[2] = (u32)(ga & 0xffffffffu);
  g[3] = (u32)((ga >> 32) & 0x1ffffffu) | (2u << 30);
  return g;
}

// Group 1: data_size code, tensor_dim0/1, tile_dim0/1, tensor_dim0_stride.
__device__ __forceinline__ v8i tdm_g1_2d(u32 dsz, u32 td0, u32 td1,
                                         u32 t0, u32 t1, uint64_t s0) {
  v8i g = {};
  g[0] = (int)(dsz << 16);                                      // data_size
  g[1] = (int)((td0 & 0xffffu) << 16);                          // td0 lo16
  g[2] = (int)(((td0 >> 16) & 0xffffu) | ((td1 & 0xffffu) << 16));
  g[3] = (int)(((td1 >> 16) & 0xffffu) | ((t0 & 0xffffu) << 16)); // tile_dim0
  g[4] = (int)(t1 & 0xffffu);                                   // tile_dim1
  g[5] = (int)(s0 & 0xffffffffu);                               // stride0 lo
  g[6] = (int)((s0 >> 32) & 0xffffu);                           // stride0 hi
  g[7] = 0;
  return g;
}

__device__ __forceinline__ u32 lds_off(const void* p) {
  return (u32)(uintptr_t)p;   // LDS aperture: addr[31:0] == LDS byte offset
}

// 1D contiguous copy of `nelem` bf16 elements (data_size code 1 = 2 bytes).
__device__ __forceinline__ void tdm_copy_1d_bf16(const void* gsrc, void* dst,
                                                 u32 nelem) {
  v4i z4 = {};
  v8i z8 = {};
  __builtin_amdgcn_tensor_load_to_lds(
      tdm_g0(lds_off(dst), gsrc),
      tdm_g1_2d(1u, nelem, 1u, nelem, 1u, (uint64_t)nelem), z4, z4, z8, 0);
}

// 2D strided copy: t0 x t1 f32 tile, row pitch `pitch` elements (code 2=4B).
__device__ __forceinline__ void tdm_copy_2d_f32(const void* gsrc, void* dst,
                                                u32 t0, u32 t1, u32 pitch) {
  v4i z4 = {};
  v8i z8 = {};
  __builtin_amdgcn_tensor_load_to_lds(
      tdm_g0(lds_off(dst), gsrc),
      tdm_g1_2d(2u, t0, t1, t0, t1, (uint64_t)pitch), z4, z4, z8, 0);
}
#endif  // USE_TDM

// ---------------------------------------------------------------------------
// Kernel 1: Q/K/V projections.  One wave -> 16 seq rows x 64 cols for one
// (b,h).  q = xh @ Wq^T  (weights shared across heads).
// ---------------------------------------------------------------------------
__global__ __launch_bounds__(256) void qkv_proj_kernel(
    const float* __restrict__ x,  const float* __restrict__ Wq,
    const float* __restrict__ Wk, const float* __restrict__ Wv,
    bf16* __restrict__ Q, bf16* __restrict__ K, bf16* __restrict__ V) {
  int wave = (int)((blockIdx.x * blockDim.x + threadIdx.x) >> 5);
  const int tiles_per_bh = S_ / 16;
  int bh = wave / tiles_per_bh;
  int s0 = (wave % tiles_per_bh) * 16;
  int b = bh / H_, h = bh % H_;
  const float* xrow = x + ((size_t)(b * S_ + s0)) * E_ + h * D_;

  v16bf a0 = load_a_f32(xrow, E_);
  v16bf a1 = load_a_f32(xrow + 32, E_);

  int l = lane_id();
  int hi  = (l >> 4) << 3;
  int col = l & 15;

  const float* Ws[3] = {Wq, Wk, Wv};
  bf16*        Os[3] = {Q, K, V};
#pragma unroll
  for (int w = 0; w < 3; ++w) {
    const float* W = Ws[w];
    bf16* O = Os[w] + ((size_t)bh * S_ + s0) * D_;
#pragma unroll
    for (int n0 = 0; n0 < D_; n0 += 16) {
      v8f acc = {};
      v16bf b0 = load_bt_f32(W + n0 * D_, D_);
      v16bf b1 = load_bt_f32(W + n0 * D_ + 32, D_);
      acc = __builtin_amdgcn_wmma_f32_16x16x32_bf16(false, a0, false, b0,
                                                    (short)0, acc, false, false);
      acc = __builtin_amdgcn_wmma_f32_16x16x32_bf16(false, a1, false, b1,
                                                    (short)0, acc, false, false);
#pragma unroll
      for (int r = 0; r < 8; ++r)
        O[(size_t)(r + hi) * D_ + n0 + col] = (__bf16)acc[r];
    }
  }
}

// ---------------------------------------------------------------------------
// Kernel 2: flash attention.  Block = 8 waves sharing one (b,h); K/V 32-key
// tiles are TDM-DMA'd into double-buffered LDS (wave 0 issues, TENSORcnt
// pipelined one tile ahead).  Online softmax in f32 on the WMMA C layout;
// P rounded to bf16 and re-laid-out through LDS into the A layout for P@V.
// ---------------------------------------------------------------------------
__global__ __launch_bounds__(256) void flash_attn_kernel(
    const bf16* __restrict__ Q, const bf16* __restrict__ K,
    const bf16* __restrict__ V, bf16* __restrict__ AO) {
  __shared__ bf16 Ks[2][KT * D_];         // 2 x 4 KB
  __shared__ bf16 Vs[2][KT * D_];         // 2 x 4 KB
  __shared__ bf16 Ps[WAVES][16 * KT];     // per-wave P tile (8 KB)

  int wave = (int)(threadIdx.x >> 5);
  int l = lane_id();
  const int blocks_per_bh = S_ / (16 * WAVES);  // 16
  int bh = (int)blockIdx.x / blocks_per_bh;
  int s0 = ((int)blockIdx.x % blocks_per_bh) * (16 * WAVES) + wave * 16;

  const bf16* Qp = Q + ((size_t)bh * S_ + s0) * D_;
  const bf16* Kp = K + (size_t)bh * S_ * D_;
  const bf16* Vp = V + (size_t)bh * S_ * D_;

  v16bf qa0 = load_a_bf16(Qp, D_);
  v16bf qa1 = load_a_bf16(Qp + 32, D_);

  v8f o[4] = {};
  float m_i[8], l_i[8];
#pragma unroll
  for (int r = 0; r < 8; ++r) { m_i[r] = -1e30f; l_i[r] = 0.0f; }
  const float scale = 0.125f;    // 1/sqrt(D)

  int hi  = (l >> 4) << 3;
  int col = l & 15;

  const int NIT = S_ / KT;       // 64 iterations

#if USE_TDM
  if (wave == 0) {               // prologue: stage tile 0
    tdm_copy_1d_bf16(Kp, &Ks[0][0], KT * D_);
    tdm_copy_1d_bf16(Vp, &Vs[0][0], KT * D_);
  }
#endif

  for (int it = 0; it < NIT; ++it) {
    int buf = it & 1;
#if USE_TDM
    __syncthreads();             // all waves done reading buf^1 (iter it-1)
    if (wave == 0) {
      if (it + 1 < NIT) {
        const size_t nxt = (size_t)(it + 1) * KT * D_;
        tdm_copy_1d_bf16(Kp + nxt, &Ks[buf ^ 1][0], KT * D_);
        tdm_copy_1d_bf16(Vp + nxt, &Vs[buf ^ 1][0], KT * D_);
        __builtin_amdgcn_s_wait_tensorcnt(2);   // tile `it` pair has landed
      } else {
        __builtin_amdgcn_s_wait_tensorcnt(0);
      }
    }
    __syncthreads();             // tile `it` visible to all waves
#else
    __syncthreads();
    for (int t = (int)threadIdx.x; t < KT * D_; t += (int)blockDim.x) {
      Ks[buf][t] = Kp[(size_t)it * KT * D_ + t];
      Vs[buf][t] = Vp[(size_t)it * KT * D_ + t];
    }
    __syncthreads();
#endif
    const bf16* Kb = &Ks[buf][0];
    const bf16* Vb = &Vs[buf][0];

    // S = Q K^T over 32 keys: two 16x16 C tiles, 2 K-steps each (d = 64).
    v8f sv0 = {}, sv1 = {};
    {
      v16bf b00 = load_bt_bf16(Kb, D_);
      v16bf b01 = load_bt_bf16(Kb + 32, D_);
      sv0 = __builtin_amdgcn_wmma_f32_16x16x32_bf16(false, qa0, false, b00,
                                                    (short)0, sv0, false, false);
      sv0 = __builtin_amdgcn_wmma_f32_16x16x32_bf16(false, qa1, false, b01,
                                                    (short)0, sv0, false, false);
      v16bf b10 = load_bt_bf16(Kb + 16 * D_, D_);
      v16bf b11 = load_bt_bf16(Kb + 16 * D_ + 32, D_);
      sv1 = __builtin_amdgcn_wmma_f32_16x16x32_bf16(false, qa0, false, b10,
                                                    (short)0, sv1, false, false);
      sv1 = __builtin_amdgcn_wmma_f32_16x16x32_bf16(false, qa1, false, b11,
                                                    (short)0, sv1, false, false);
    }

    // Online softmax.  Row r+hi lives across the 16 lanes of this half-wave.
    float sc[8];
#pragma unroll
    for (int r = 0; r < 8; ++r) {
      float a = sv0[r] * scale;
      float b = sv1[r] * scale;
      float mx = fmaxf(a, b);
      for (int mask = 1; mask < 16; mask <<= 1)
        mx = fmaxf(mx, __shfl_xor(mx, mask, 32));
      float mnew = fmaxf(m_i[r], mx);
      sc[r] = __expf(m_i[r] - mnew);
      m_i[r] = mnew;
      float p0 = __expf(a - mnew);
      float p1 = __expf(b - mnew);
      sv0[r] = p0; sv1[r] = p1;
      float ps = p0 + p1;
      for (int mask = 1; mask < 16; mask <<= 1)
        ps += __shfl_xor(ps, mask, 32);
      l_i[r] = l_i[r] * sc[r] + ps;
#pragma unroll
      for (int nt = 0; nt < 4; ++nt) o[nt][r] *= sc[r];
    }

    // Stage P (16x32) to LDS as bf16 row-major, reload in A layout.
    bf16* P = &Ps[wave][0];
#pragma unroll
    for (int r = 0; r < 8; ++r) {
      P[(r + hi) * KT + col]      = (__bf16)sv0[r];
      P[(r + hi) * KT + 16 + col] = (__bf16)sv1[r];
    }
    __builtin_amdgcn_wave_barrier();   // wave-coherent LDS exchange
    v16bf pa = load_a_bf16(P, KT);

#pragma unroll
    for (int nt = 0; nt < 4; ++nt) {
      v16bf bv = load_b_bf16(Vb + nt * 16, D_);
      o[nt] = __builtin_amdgcn_wmma_f32_16x16x32_bf16(false, pa, false, bv,
                                                      (short)0, o[nt], false, false);
    }
  }

  // Finalize 1/l and store bf16 into (B, S, E) with heads concatenated.
  int b_ = bh / H_, h = bh % H_;
  bf16* Op = AO + ((size_t)(b_ * S_ + s0)) * E_ + h * D_;
#pragma unroll
  for (int r = 0; r < 8; ++r) {
    float inv = 1.0f / l_i[r];
#pragma unroll
    for (int nt = 0; nt < 4; ++nt)
      Op[(size_t)(r + hi) * E_ + nt * 16 + col] = (__bf16)(o[nt][r] * inv);
  }
}

// ---------------------------------------------------------------------------
// Kernel 3: output projection  out = AO @ Wo^T + bo, f32 out.
// Block = 8 waves -> 128 rows x 64 cols.  Wo 64x64 f32 chunks are TDM-DMA'd
// (2D strided descriptor, row pitch E) into double-buffered LDS.
// ---------------------------------------------------------------------------
__global__ __launch_bounds__(256) void out_proj_kernel(
    const bf16* __restrict__ A, const float* __restrict__ Wo,
    const float* __restrict__ bo, float* __restrict__ out) {
  __shared__ float Wf[2][64 * 64];   // 2 x 16 KB, [n][k] chunks

  int wave = (int)(threadIdx.x >> 5);
  int l = lane_id();
  int m0 = (int)blockIdx.x * 128 + wave * 16;
  int n0 = (int)blockIdx.y * 64;

  const int NCH = E_ / 64;           // 16 k-chunks

#if USE_TDM
  if (wave == 0)
    tdm_copy_2d_f32(&Wo[(size_t)n0 * E_], &Wf[0][0], 64u, 64u, (u32)E_);
#endif

  v8f acc[4] = {};
  for (int c = 0; c < NCH; ++c) {
    int buf = c & 1;
    int k0 = c * 64;
#if USE_TDM
    __syncthreads();
    if (wave == 0) {
      if (c + 1 < NCH) {
        tdm_copy_2d_f32(&Wo[(size_t)n0 * E_ + (size_t)(c + 1) * 64],
                        &Wf[buf ^ 1][0], 64u, 64u, (u32)E_);
        __builtin_amdgcn_s_wait_tensorcnt(1);
      } else {
        __builtin_amdgcn_s_wait_tensorcnt(0);
      }
    }
    __syncthreads();
#else
    __syncthreads();
    for (int t = (int)threadIdx.x; t < 64 * 64; t += (int)blockDim.x) {
      int n = t >> 6, k = t & 63;
      Wf[buf][t] = Wo[(size_t)(n0 + n) * E_ + k0 + k];
    }
    __syncthreads();
#endif

    v16bf a0 = load_a_bf16(A + (size_t)m0 * E_ + k0, E_);
    v16bf a1 = load_a_bf16(A + (size_t)m0 * E_ + k0 + 32, E_);
#pragma unroll
    for (int nt = 0; nt < 4; ++nt) {
      v16bf b0 = load_bt_f32(&Wf[buf][nt * 16 * 64], 64);
      v16bf b1 = load_bt_f32(&Wf[buf][nt * 16 * 64 + 32], 64);
      acc[nt] = __builtin_amdgcn_wmma_f32_16x16x32_bf16(false, a0, false, b0,
                                                        (short)0, acc[nt], false, false);
      acc[nt] = __builtin_amdgcn_wmma_f32_16x16x32_bf16(false, a1, false, b1,
                                                        (short)0, acc[nt], false, false);
    }
  }

  int hi  = (l >> 4) << 3;
  int col = l & 15;
#pragma unroll
  for (int nt = 0; nt < 4; ++nt) {
    float bias = bo[n0 + nt * 16 + col];
#pragma unroll
    for (int r = 0; r < 8; ++r)
      out[(size_t)(m0 + r + hi) * E_ + n0 + nt * 16 + col] = acc[nt][r] + bias;
  }
}

// ---------------------------------------------------------------------------
extern "C" void kernel_launch(void* const* d_in, const int* in_sizes, int n_in,
                              void* d_out, int out_size, void* d_ws, size_t ws_size,
                              hipStream_t stream) {
  (void)in_sizes; (void)n_in; (void)out_size; (void)ws_size;
  const float* x  = (const float*)d_in[0];
  const float* Wq = (const float*)d_in[1];
  const float* Wk = (const float*)d_in[2];
  const float* Wv = (const float*)d_in[3];
  const float* Wo = (const float*)d_in[4];
  const float* bo = (const float*)d_in[5];
  float* out = (float*)d_out;

  const size_t qkv_elems = (size_t)B_ * H_ * S_ * D_;   // 4,194,304
  bf16* Q  = (bf16*)d_ws;
  bf16* K  = Q + qkv_elems;
  bf16* V  = K + qkv_elems;
  bf16* AO = V + qkv_elems;                             // B*S*E bf16

  qkv_proj_kernel<<<dim3(512), dim3(256), 0, stream>>>(x, Wq, Wk, Wv, Q, K, V);
  flash_attn_kernel<<<dim3(512), dim3(256), 0, stream>>>(Q, K, V, AO);
  out_proj_kernel<<<dim3(32, 16), dim3(256), 0, stream>>>(AO, Wo, bo, out);
}
